// DecoderRNN_17755394801737
// MI455X (gfx1250) — compile-verified, matrix operand order checked by reading, not run on hardware
//
#include <hip/hip_runtime.h>
#include <hip/hip_bf16.h>

// ---------------------------------------------------------------------------
// DecoderRNN (show-attend-tell LSTM) for MI455X / gfx1250, wave32 + WMMA bf16.
// B=64, N=196, VD=ED=HD=512, V=10000, T=20
// ---------------------------------------------------------------------------

#define B_   64
#define N_   196
#define VD_  512
#define HD_  512
#define ED_  512
#define V_   10000
#define T_   20
#define KG_  1536          // K for gates GEMM: VD + ED + HD
#define NG_  2048          // 4*HD

typedef __attribute__((ext_vector_type(16))) __bf16 v16bf;
typedef __attribute__((ext_vector_type(8)))  __bf16 v8bf;
typedef __attribute__((ext_vector_type(8)))  float  v8f;

// --- WMMA operand loader: 16-bit A/Bt layout (lane = row, K-chunks at
//     kbase = 8*(lane>>4) and kbase+16, each 8 contiguous bf16 = 16 bytes) ---
__device__ __forceinline__ v16bf load_ab16(const __bf16* p) {
  v8bf lo = *reinterpret_cast<const v8bf*>(p);
  v8bf hi = *reinterpret_cast<const v8bf*>(p + 16);
  v16bf r;
#pragma unroll
  for (int i = 0; i < 8; ++i) { r[i] = lo[i]; r[i + 8] = hi[i]; }
  return r;
}

__device__ __forceinline__ v8f wmma_bf16(v16bf a, v16bf b, v8f c) {
  return __builtin_amdgcn_wmma_f32_16x16x32_bf16(
      /*neg_a=*/false, a, /*neg_b=*/false, b,
      /*c_mod=*/(short)0, c, /*reuse_a=*/false, /*reuse_b=*/false);
}

// ---------------------------------------------------------------------------
// Generic bf16 tile GEMM:  C[M][N] = A[M][K] * W[N][K]^T (+ bias[N])
// One wave per 16x16 output tile; K stepped by 32 via v_wmma_f32_16x16x32_bf16
// ---------------------------------------------------------------------------
__global__ __launch_bounds__(256)
void k_wmma_gemm(const __bf16* __restrict__ A, const __bf16* __restrict__ W,
                 float* __restrict__ C, const float* __restrict__ bias,
                 int M, int N, int K) {
  int wave = (blockIdx.x * blockDim.x + threadIdx.x) >> 5;
  int lane = threadIdx.x & 31;
  int ntn  = N >> 4;
  int tm   = (wave / ntn) << 4;
  int tn   = (wave % ntn) << 4;
  if (tm >= M) return;                       // uniform per wave

  int rsel = lane & 15;                      // row within tile (A) / col (W^T)
  int hh   = lane >> 4;                      // K half select
  const __bf16* Ap = A + (size_t)(tm + rsel) * K + hh * 8;
  const __bf16* Wp = W + (size_t)(tn + rsel) * K + hh * 8;

  v8f acc = {};
  for (int k = 0; k < K; k += 32) {
    __builtin_prefetch(Wp + k + 256, 0, 0);  // global_prefetch_b8 on weight stream
    v16bf a = load_ab16(Ap + k);
    v16bf b = load_ab16(Wp + k);
    acc = wmma_bf16(a, b, acc);
  }

  int col = tn + rsel;
  float bv = bias ? bias[col] : 0.0f;
  float* Crow = C + (size_t)(tm + hh * 8) * N + col;   // VGPR r -> row tm+8*hh+r
#pragma unroll
  for (int r = 0; r < 8; ++r) Crow[(size_t)r * N] = acc[r] + bv;
}

// ---------------------------------------------------------------------------
// Vocab GEMM:  words[64][10000] = h_bf[64][512] . Wout_bf[10000][512]^T + b_out
// A (64x512 bf16 = 64KB) is staged into LDS once per block using the gfx1250
// async global->LDS engine (ASYNCcnt), then each wave computes 1 M-tile x
// 5 N-tiles, reusing the LDS A-fragment across 5 WMMAs per K-step while W_out
// streams from L2 via global_load_b128.
// ---------------------------------------------------------------------------
__global__ __launch_bounds__(256)
void k_wmma_vocab(const __bf16* __restrict__ Abf, const __bf16* __restrict__ W,
                  float* __restrict__ C, const float* __restrict__ bias) {
  __shared__ __bf16 As[B_ * HD_];            // 64 KB of 320 KB/WGP
  int tid = threadIdx.x;

  // Async copy global -> LDS: 65536 B in 16B chunks, 256 threads, 16 iters.
  {
    const char* g = (const char*)Abf;
    char*       l = (char*)As;
#pragma unroll 4
    for (int i = tid * 16; i < B_ * HD_ * 2; i += 256 * 16) {
      unsigned int  lds_off = (unsigned int)(unsigned long long)(l + i);
      unsigned long long ga = (unsigned long long)(g + i);
      asm volatile("global_load_async_to_lds_b128 %0, %1, off"
                   :: "v"(lds_off), "v"(ga) : "memory");
    }
    asm volatile("s_wait_asynccnt 0x0" ::: "memory");
    __syncthreads();
  }

  int wave = (blockIdx.x * blockDim.x + tid) >> 5;
  int lane = tid & 31;
  if (wave >= 4 * 125) return;               // 4 M-tiles x 125 groups of 5 N-tiles

  int tm   = (wave / 125) << 4;
  int ng   = (wave % 125) * 5;               // first N-tile of this wave's group
  int rsel = lane & 15;
  int hh   = lane >> 4;

  const __bf16* Ap = As + (tm + rsel) * HD_ + hh * 8;               // LDS
  const __bf16* Wp = W + (size_t)(ng * 16 + rsel) * HD_ + hh * 8;   // global

  v8f acc[5] = {};
  for (int k = 0; k < HD_; k += 32) {
    v16bf a = load_ab16(Ap + k);             // ds_load_b128 x2
#pragma unroll
    for (int j = 0; j < 5; ++j) {
      v16bf b = load_ab16(Wp + (size_t)j * 16 * HD_ + k);
      acc[j] = wmma_bf16(a, b, acc[j]);
    }
  }

#pragma unroll
  for (int j = 0; j < 5; ++j) {
    int col = (ng + j) * 16 + rsel;
    float bv = bias[col];
    float* Crow = C + (size_t)(tm + hh * 8) * V_ + col;
#pragma unroll
    for (int r = 0; r < 8; ++r) Crow[(size_t)r * V_] = acc[j][r] + bv;
  }
}

// ---------------------------------------------------------------------------
// fp32 -> bf16 convert
// ---------------------------------------------------------------------------
__global__ void k_cvt_bf16(const float* __restrict__ in, __bf16* __restrict__ out,
                           size_t n) {
  size_t i = (size_t)blockIdx.x * blockDim.x + threadIdx.x;
  if (i < n) out[i] = (__bf16)in[i];
}

// Pack Wg[2048][1536] = [ W_ih[2048][1024] | W_hh[2048][512] ] as bf16
__global__ void k_pack_wg(const float* __restrict__ W_ih,
                          const float* __restrict__ W_hh,
                          __bf16* __restrict__ Wg) {
  size_t i = (size_t)blockIdx.x * blockDim.x + threadIdx.x;
  if (i >= (size_t)NG_ * KG_) return;
  int row = (int)(i / KG_), col = (int)(i % KG_);
  float v = (col < VD_ + ED_) ? W_ih[(size_t)row * (VD_ + ED_) + col]
                              : W_hh[(size_t)row * HD_ + (col - VD_ - ED_)];
  Wg[i] = (__bf16)v;
}

// ---------------------------------------------------------------------------
// Per-batch prep: fsum[b][v] = sum_n features,  att_v[b][n] = f.W_av + b_av
// ---------------------------------------------------------------------------
__global__ __launch_bounds__(256)
void k_prep(const float* __restrict__ features, const float* __restrict__ W_av,
            const float* __restrict__ b_av, float* __restrict__ fsum,
            float* __restrict__ att_v) {
  int b = blockIdx.x, tid = threadIdx.x;
  const float* fb = features + (size_t)b * N_ * VD_;
  for (int v = tid; v < VD_; v += 256) {
    float s = 0.f;
    for (int n = 0; n < N_; ++n) s += fb[(size_t)n * VD_ + v];
    fsum[b * VD_ + v] = s;
  }
  for (int n = tid; n < N_; n += 256) {
    const float* row = fb + (size_t)n * VD_;
    float s = 0.f;
    for (int v = 0; v < VD_; ++v) s += row[v] * W_av[v];
    att_v[b * N_ + n] = s + b_av[0];
  }
}

// h0/c0: fsum @ W_init^T / N   (small fp32 GEMM, VALU is fine)
__global__ __launch_bounds__(256)
void k_h0c0(const float* __restrict__ fsum, const float* __restrict__ Wih,
            const float* __restrict__ Wic, float* __restrict__ h,
            float* __restrict__ c) {
  int b = blockIdx.x, tid = threadIdx.x;
  const float* fb = fsum + b * VD_;
  for (int j = tid; j < HD_; j += 256) {
    const float* wh = Wih + (size_t)j * VD_;
    const float* wc = Wic + (size_t)j * VD_;
    float sh = 0.f, sc = 0.f;
    for (int v = 0; v < VD_; ++v) { sh += fb[v] * wh[v]; sc += fb[v] * wc[v]; }
    h[b * HD_ + j] = sh * (1.0f / N_);
    c[b * HD_ + j] = sc * (1.0f / N_);
  }
}

// ---------------------------------------------------------------------------
// Attention + context + pack x_bf[b] = [bf16(ctx) | emb_bf[word] | bf16(h)]
// ---------------------------------------------------------------------------
__global__ __launch_bounds__(256)
void k_attn_ctx(const float* __restrict__ features, const float* __restrict__ att_v,
                const float* __restrict__ h, const float* __restrict__ W_ah,
                const float* __restrict__ b_ah, const __bf16* __restrict__ emb_bf,
                const int* __restrict__ captions, int t,
                __bf16* __restrict__ x_bf) {
  int b = blockIdx.x, tid = threadIdx.x;
  __shared__ float red[256];
  __shared__ float w[N_];
  const float* hb = h + b * HD_;

  // att_h = h . W_ah + b_ah
  float s = 0.f;
  for (int v = tid; v < HD_; v += 256) s += hb[v] * W_ah[v];
  red[tid] = s; __syncthreads();
  for (int off = 128; off > 0; off >>= 1) {
    if (tid < off) red[tid] += red[tid + off];
    __syncthreads();
  }
  float att_h = red[0] + b_ah[0];
  __syncthreads();

  // softmax over n of (att_v + att_h)
  const float* av = att_v + b * N_;
  float m = -1e30f;
  for (int n = tid; n < N_; n += 256) { float sc = av[n] + att_h; w[n] = sc; m = fmaxf(m, sc); }
  red[tid] = m; __syncthreads();
  for (int off = 128; off > 0; off >>= 1) {
    if (tid < off) red[tid] = fmaxf(red[tid], red[tid + off]);
    __syncthreads();
  }
  float mx = red[0]; __syncthreads();
  float e = 0.f;
  for (int n = tid; n < N_; n += 256) { float ex = __expf(w[n] - mx); w[n] = ex; e += ex; }
  red[tid] = e; __syncthreads();
  for (int off = 128; off > 0; off >>= 1) {
    if (tid < off) red[tid] += red[tid + off];
    __syncthreads();
  }
  float inv = 1.0f / red[0];
  __syncthreads();

  // ctx[v] = sum_n features[b][n][v] * attn[n];  pack x_bf
  const float* fb = features + (size_t)b * N_ * VD_;
  __bf16* xb = x_bf + (size_t)b * KG_;
  for (int v = tid; v < VD_; v += 256) {
    float acc = 0.f;
    for (int n = 0; n < N_; ++n) acc += fb[(size_t)n * VD_ + v] * (w[n] * inv);
    xb[v] = (__bf16)acc;
  }
  int word = captions[b * T_ + t];
  const __bf16* er = emb_bf + (size_t)word * ED_;
  for (int v = tid; v < ED_; v += 256) xb[VD_ + v] = er[v];
  for (int v = tid; v < HD_; v += 256) xb[VD_ + ED_ + v] = (__bf16)hb[v];
}

// ---------------------------------------------------------------------------
// LSTM pointwise: gates [64][2048] (i|f|g|o) -> h, c, h_bf
// ---------------------------------------------------------------------------
__global__ __launch_bounds__(256)
void k_lstm(const float* __restrict__ gates, const float* __restrict__ b_ih,
            const float* __restrict__ b_hh, float* __restrict__ h,
            float* __restrict__ c, __bf16* __restrict__ h_bf) {
  int idx = blockIdx.x * blockDim.x + threadIdx.x;
  if (idx >= B_ * HD_) return;
  int b = idx >> 9, j = idx & (HD_ - 1);
  const float* g = gates + (size_t)b * NG_;
  float iv = g[j]            + b_ih[j]            + b_hh[j];
  float fv = g[j + HD_]      + b_ih[j + HD_]      + b_hh[j + HD_];
  float gv = g[j + 2 * HD_]  + b_ih[j + 2 * HD_]  + b_hh[j + 2 * HD_];
  float ov = g[j + 3 * HD_]  + b_ih[j + 3 * HD_]  + b_hh[j + 3 * HD_];
  float si = 1.f / (1.f + __expf(-iv));
  float sf = 1.f / (1.f + __expf(-fv));
  float so = 1.f / (1.f + __expf(-ov));
  float cn = sf * c[idx] + si * tanhf(gv);
  float hn = so * tanhf(cn);
  c[idx] = cn;
  h[idx] = hn;
  h_bf[idx] = (__bf16)hn;
}

// ---------------------------------------------------------------------------
// log_softmax + softmax over V=10000 per (t,b); writes both output halves
// ---------------------------------------------------------------------------
__global__ __launch_bounds__(256)
void k_softmax(const float* __restrict__ words, float* __restrict__ out, int t) {
  int b = blockIdx.x, tid = threadIdx.x;
  __shared__ float red[256];
  const float* row = words + (size_t)b * V_;
  float m = -1e30f;
  for (int v = tid; v < V_; v += 256) m = fmaxf(m, row[v]);
  red[tid] = m; __syncthreads();
  for (int off = 128; off > 0; off >>= 1) {
    if (tid < off) red[tid] = fmaxf(red[tid], red[tid + off]);
    __syncthreads();
  }
  float mx = red[0]; __syncthreads();
  float s = 0.f;
  for (int v = tid; v < V_; v += 256) s += __expf(row[v] - mx);
  red[tid] = s; __syncthreads();
  for (int off = 128; off > 0; off >>= 1) {
    if (tid < off) red[tid] += red[tid + off];
    __syncthreads();
  }
  float sum = red[0];
  float lse = __logf(sum);
  float inv = 1.0f / sum;
  float* ls = out + ((size_t)t * B_ + b) * V_;
  float* sm = out + (size_t)T_ * B_ * V_ + ((size_t)t * B_ + b) * V_;
  for (int v = tid; v < V_; v += 256) {
    float z = row[v] - mx;
    ls[v] = z - lse;
    sm[v] = __expf(z) * inv;
  }
}

// ---------------------------------------------------------------------------
extern "C" void kernel_launch(void* const* d_in, const int* in_sizes, int n_in,
                              void* d_out, int out_size, void* d_ws, size_t ws_size,
                              hipStream_t stream) {
  const float* features = (const float*)d_in[0];
  const int*   captions = (const int*)  d_in[1];
  const float* W_init_h = (const float*)d_in[2];
  const float* W_init_c = (const float*)d_in[3];
  const float* W_av     = (const float*)d_in[4];
  const float* b_av     = (const float*)d_in[5];
  const float* W_ah     = (const float*)d_in[6];
  const float* b_ah     = (const float*)d_in[7];
  const float* embed    = (const float*)d_in[8];
  const float* W_ih     = (const float*)d_in[9];
  const float* W_hh     = (const float*)d_in[10];
  const float* b_ih     = (const float*)d_in[11];
  const float* b_hh     = (const float*)d_in[12];
  const float* W_out    = (const float*)d_in[13];
  const float* b_out    = (const float*)d_in[14];
  float* out = (float*)d_out;

  // Bump-allocate workspace (256B aligned).  Total ~30.5 MB.
  char* p = (char*)d_ws;
  auto alloc = [&](size_t bytes) -> void* {
    void* r = (void*)p;
    p += (bytes + 255) & ~(size_t)255;
    return r;
  };
  __bf16* Wg_bf   = (__bf16*)alloc((size_t)NG_ * KG_ * 2);      // 6.0 MB
  __bf16* Wout_bf = (__bf16*)alloc((size_t)V_ * HD_ * 2);       // 9.8 MB
  __bf16* emb_bf  = (__bf16*)alloc((size_t)V_ * ED_ * 2);       // 9.8 MB
  float*  fsum    = (float*) alloc((size_t)B_ * VD_ * 4);
  float*  att_v   = (float*) alloc((size_t)B_ * N_ * 4);
  float*  h       = (float*) alloc((size_t)B_ * HD_ * 4);
  float*  c       = (float*) alloc((size_t)B_ * HD_ * 4);
  __bf16* h_bf    = (__bf16*)alloc((size_t)B_ * HD_ * 2);
  __bf16* x_bf    = (__bf16*)alloc((size_t)B_ * KG_ * 2);
  float*  gates   = (float*) alloc((size_t)B_ * NG_ * 4);
  float*  words   = (float*) alloc((size_t)B_ * V_ * 4);

  // One-time (per launch) weight conversion to bf16
  {
    size_t n = (size_t)V_ * ED_;
    k_cvt_bf16<<<dim3((unsigned)((n + 255) / 256)), dim3(256), 0, stream>>>(embed, emb_bf, n);
    n = (size_t)V_ * HD_;
    k_cvt_bf16<<<dim3((unsigned)((n + 255) / 256)), dim3(256), 0, stream>>>(W_out, Wout_bf, n);
    n = (size_t)NG_ * KG_;
    k_pack_wg<<<dim3((unsigned)((n + 255) / 256)), dim3(256), 0, stream>>>(W_ih, W_hh, Wg_bf);
  }

  // Init: feature sums, attention logits, h0/c0
  k_prep<<<dim3(B_), dim3(256), 0, stream>>>(features, W_av, b_av, fsum, att_v);
  k_h0c0<<<dim3(B_), dim3(256), 0, stream>>>(fsum, W_init_h, W_init_c, h, c);

  // Recurrence
  for (int t = 0; t < T_; ++t) {
    k_attn_ctx<<<dim3(B_), dim3(256), 0, stream>>>(
        features, att_v, h, W_ah, b_ah, emb_bf, captions, t, x_bf);

    // gates = x_bf[64][1536] . Wg_bf[2048][1536]^T : 4*128 = 512 tiles -> 64 blocks
    k_wmma_gemm<<<dim3(64), dim3(256), 0, stream>>>(
        x_bf, Wg_bf, gates, nullptr, B_, NG_, KG_);

    k_lstm<<<dim3((B_ * HD_ + 255) / 256), dim3(256), 0, stream>>>(
        gates, b_ih, b_hh, h, c, h_bf);

    // words = h_bf . Wout_bf^T + b_out : 500 waves (1 M-tile x 5 N-tiles each)
    k_wmma_vocab<<<dim3((500 + 7) / 8), dim3(256), 0, stream>>>(
        h_bf, Wout_bf, words, b_out);

    k_softmax<<<dim3(B_), dim3(256), 0, stream>>>(words, out, t);
  }
}
